// STFT_37031208026532
// MI455X (gfx1250) — matrix operand order repaired and checked by reading, not compiled, not run on hardware
//
#include <hip/hip_runtime.h>
#include <hip/hip_bf16.h>
#include <math.h>

typedef _Float16 v16h __attribute__((ext_vector_type(16)));
typedef _Float16 v8h  __attribute__((ext_vector_type(8)));
typedef _Float16 v4h  __attribute__((ext_vector_type(4)));
typedef float    v8f  __attribute__((ext_vector_type(8)));
typedef float    v4f  __attribute__((ext_vector_type(4)));

#define FILTER_LEN 800
#define HOP        200
#define CUTOFF     401
#define BATCH      32
#define SIG_LEN    320000
#define N_FRAMES   1601
#define M_TILES    51          // ceil(802/16)
#define K_CHUNKS   25          // 800/32
#define N_TILES    101         // ceil(1601/16)
#define M_GROUPS   13          // ceil(51/4)
#define F_CHUNKS   8           // frame-tile chunks
#define TPC        13          // tiles per chunk (8*13 = 104 >= 101)
#define FRAG_ELEMS (M_TILES * K_CHUNKS * 32 * 16)   // 652800 f16 (1.3 MB)

// sched_group_barrier masks
#define SG_WMMA    0x008
#define SG_DSREAD  0x100

// ---------------------------------------------------------------------------
// Stage 0: windowed-DFT basis, pre-swizzled into the CDNA5 A-fragment layout.
// Row interleave: m = 2q -> win[k]*cos(2*pi*q*k/800), m = 2q+1 -> -win[k]*sin.
// A 16x32 layout used here (contiguous k, same as B): lane l -> M = l&15;
//   lanes 0-15 : element i -> K = i
//   lanes 16-31: element i -> K = 16 + i
// Flat: ((mt*25 + kc)*32 + lane)*16 + i -> 32 contiguous bytes per lane.
// ---------------------------------------------------------------------------
__global__ void stft_basis_frag(_Float16* __restrict__ aFrag) {
    int idx = blockIdx.x * blockDim.x + threadIdx.x;
    if (idx >= FRAG_ELEMS) return;
    int i    = idx & 15;
    int lane = (idx >> 4) & 31;
    int kc   = (idx >> 9) % K_CHUNKS;
    int mt   = idx / (K_CHUNKS << 9);
    int m = mt * 16 + (lane & 15);
    int k = kc * 32 + ((lane >> 4) << 4) + i;
    float v = 0.0f;
    int q = m >> 1;
    if (q < CUTOFF) {
        const float w = 6.283185307179586476925f / 800.0f;
        float win = 0.5f - 0.5f * cosf(w * (float)k);
        int r = (q * k) % 800;                 // exact angle reduction
        float ang = w * (float)r;
        float tri = (m & 1) ? -sinf(ang) : cosf(ang);
        v = win * tri;
    }
    aFrag[idx] = (_Float16)v;
}

// ---------------------------------------------------------------------------
// Stage 1: weights-stationary batched WMMA GEMM.
// Wave w owns basis M-tile mt = 4*blockIdx.z + w: all 25 A fragments (full K)
// stay resident in VGPRs. The block sweeps TPC frame tiles of one batch:
// per tile the 3800-sample reflect-padded window is staged into LDS once,
// then each wave runs a 2-deep software-pipelined ds_load_b128 -> WMMA chain,
// with sched_group_barrier pinning the [WMMA, 2xDS-read] issue pattern.
// Inner loop: zero global loads.
// ---------------------------------------------------------------------------
__global__ __launch_bounds__(128, 1)
void stft_wmma(const float* __restrict__ x,
               const _Float16* __restrict__ aFrag,
               float* __restrict__ out) {
    __shared__ __align__(16) _Float16 sx[3808];

    const int chunk = blockIdx.x;      // frame-tile chunk
    const int b     = blockIdx.y;      // batch
    const int g     = blockIdx.z;      // M-tile group
    const int tid   = threadIdx.x;
    const int wave  = tid >> 5;
    const int lane  = tid & 31;
    const int n     = lane & 15;       // frame column within tile
    const int hi    = lane >> 4;

    const int  mt     = g * 4 + wave;
    const bool active = (mt < M_TILES);          // wave-uniform
    const float* xb   = x + (size_t)b * SIG_LEN;

    // --- A resident: 25 fragments (200 VGPRs), loaded once from L2 ---------
    v16h afrag[K_CHUNKS];
    if (active) {
        const v16h* ap = (const v16h*)aFrag + (size_t)mt * K_CHUNKS * 32 + lane;
        #pragma unroll
        for (int kc = 0; kc < K_CHUNKS; ++kc) afrag[kc] = ap[kc * 32];
    }

    const size_t plane = (size_t)CUTOFF * N_FRAMES;
    float* magp = out + (size_t)b * plane;
    float* phap = out + (size_t)BATCH * plane + (size_t)b * plane;

    for (int tl = 0; tl < TPC; ++tl) {
        const int t = chunk * TPC + tl;
        if (t >= N_TILES) break;                 // block-uniform

        __syncthreads();                         // protect LDS reuse
        const int base = t * (16 * HOP);
        if (t >= 1 && t <= 98) {
            // Interior tile: no reflection possible -> vectorized staging.
            const float* src = xb + base - 400;  // 16B aligned
            for (int i = tid * 4; i < 3800; i += 512) {
                v4f f = *(const v4f*)(src + i);
                v4h h = { (_Float16)f.x, (_Float16)f.y,
                          (_Float16)f.z, (_Float16)f.w };
                *(v4h*)&sx[i] = h;               // ds_store_b64
            }
        } else {
            // Edge tile: scalar loads with reflect padding (pad = 400).
            for (int i = tid; i < 3800; i += 128) {
                int s = base + i - 400;
                if (s < 0) s = -s;
                if (s >= SIG_LEN) s = 2 * SIG_LEN - 2 - s;
                sx[i] = (_Float16)xb[s];
            }
        }
        __syncthreads();

        // Warm next tile's window while we compute this one.
        {
            int pi = (base + 3200) - 400 + tid * 32;
            if (pi < 0) pi = 0;
            if (pi >= SIG_LEN) pi = SIG_LEN - 1;
            __builtin_prefetch(&xb[pi], 0, 3);   // global_prefetch_b8
        }

        if (active) {
            // B 32x16 layout used here (same k order as A): lane l -> N = l&15; lanes 0-15 hold
            // K 0-15, lanes 16-31 hold K 16-31. Byte offset
            // 400*n + 64*kc + 32*hi is 16B aligned -> ds_load_b128 x2.
            auto loadB = [&](int kc) -> v16h {
                const v8h* p = (const v8h*)&sx[n * HOP + kc * 32 + hi * 16];
                v8h lo = p[0];
                v8h hh = p[1];
                return __builtin_shufflevector(lo, hh,
                           0,1,2,3,4,5,6,7,8,9,10,11,12,13,14,15);
            };

            // Explicit 2-deep software pipeline: the ds loads for chunk
            // kc+2 are issued a full WMMA ahead of their consumer.
            v8f  acc = {};
            v16h b0 = loadB(0);
            v16h b1 = loadB(1);
            #pragma unroll
            for (int kc = 0; kc < K_CHUNKS - 1; kc += 2) {
                acc = __builtin_amdgcn_wmma_f32_16x16x32_f16(
                          false, afrag[kc], false, b0,
                          (short)0, acc, false, false);
                if (kc + 2 < K_CHUNKS) b0 = loadB(kc + 2);
                acc = __builtin_amdgcn_wmma_f32_16x16x32_f16(
                          false, afrag[kc + 1], false, b1,
                          (short)0, acc, false, false);
                if (kc + 3 < K_CHUNKS) b1 = loadB(kc + 3);
            }
            acc = __builtin_amdgcn_wmma_f32_16x16x32_f16(
                      false, afrag[K_CHUNKS - 1], false, b0,
                      (short)0, acc, false, false);

            // Pin the issue order so RA cannot collapse the double buffer:
            //   [4 DS reads] then 23 x [1 WMMA, 2 DS reads] then 2 x [WMMA].
            __builtin_amdgcn_sched_group_barrier(SG_DSREAD, 4, 0);
            #pragma unroll
            for (int it = 0; it < K_CHUNKS - 2; ++it) {
                __builtin_amdgcn_sched_group_barrier(SG_WMMA,   1, 0);
                __builtin_amdgcn_sched_group_barrier(SG_DSREAD, 2, 0);
            }
            __builtin_amdgcn_sched_group_barrier(SG_WMMA, 1, 0);
            __builtin_amdgcn_sched_group_barrier(SG_WMMA, 1, 0);

            // C layout: VGPR r -> M = r (lanes 0-15) / r+8 (lanes 16-31).
            // Interleaved basis rows => (acc[2j], acc[2j+1]) = (re, im) of
            // freq = mt*8 + hi*4 + j at column `frame`.
            const int frame = t * 16 + n;
            if (frame < N_FRAMES) {
                int fb = mt * 8 + hi * 4;
                #pragma unroll
                for (int j = 0; j < 4; ++j) {
                    int freq = fb + j;
                    if (freq < CUTOFF) {
                        float re = acc[2 * j];
                        float im = acc[2 * j + 1];
                        float mag = sqrtf(re * re + im * im);
                        float ph  = atan2f(im, re);
                        size_t o = (size_t)freq * N_FRAMES + frame;
                        magp[o] = mag;
                        phap[o] = ph;
                    }
                }
            }
        }
    }
}

// ---------------------------------------------------------------------------
extern "C" void kernel_launch(void* const* d_in, const int* in_sizes, int n_in,
                              void* d_out, int out_size, void* d_ws, size_t ws_size,
                              hipStream_t stream) {
    const float* x = (const float*)d_in[0];
    float* out = (float*)d_out;
    _Float16* aFrag = (_Float16*)d_ws;   // 1,305,600 bytes of scratch

    stft_basis_frag<<<(FRAG_ELEMS + 255) / 256, 256, 0, stream>>>(aFrag);

    dim3 grid(F_CHUNKS, BATCH, M_GROUPS);
    stft_wmma<<<grid, 128, 0, stream>>>(x, aFrag, out);
}
